// RAT_28132035788999
// MI455X (gfx1250) — matrix operand order repaired and checked by reading, not compiled, not run on hardware
//
#include <hip/hip_runtime.h>
#include <math.h>

#define NN   20000
#define EE   640000
#define HSZ  256
#define NH   8
#define NL   2
#define DKH  32
#define NHS  (NN * HSZ)   // 5,120,000

typedef __bf16 v16bf __attribute__((ext_vector_type(16)));
typedef __bf16 v8bf  __attribute__((ext_vector_type(8)));
typedef float  v8f   __attribute__((ext_vector_type(8)));

union ABfrag { v16bf v; v8bf h[2]; };

// ---------------- elementwise helpers ----------------

__global__ __launch_bounds__(256) void f32_to_bf16_kernel(
    const float* __restrict__ in, __bf16* __restrict__ out, int n) {
  int i = blockIdx.x * 256 + threadIdx.x;
  if (i < n) out[i] = (__bf16)in[i];
}

__global__ __launch_bounds__(256) void zero_kernel(float* __restrict__ p, int n) {
  int i = blockIdx.x * 256 + threadIdx.x;
  if (i < n) p[i] = 0.0f;
}

// Pack row-major fp32 W[K,Ncols] into bf16 WMMA B-fragment-major layout:
// fragment f = nt*(K/32)+kt holds a 32x16 (KxN) block; within a fragment,
// lane L (0..31) stores 16 contiguous bf16: col = nt*16 + (L&15),
// krow = kt*32 + (L>>4)*16 + j  (j = 0..15).
__global__ __launch_bounds__(256) void pack_b_kernel(
    const float* __restrict__ W, __bf16* __restrict__ out, int K, int Ncols) {
  int t = blockIdx.x * 256 + threadIdx.x;       // grid sized to K*Ncols exactly
  int f    = t >> 9;
  int r    = t & 511;
  int lane = r >> 4;
  int j    = r & 15;
  int ktiles = K >> 5;
  int kt = f % ktiles;
  int nt = f / ktiles;
  int col  = nt * 16 + (lane & 15);
  int krow = kt * 32 + ((lane >> 4) << 4) + j;
  out[t] = (__bf16)W[(size_t)krow * Ncols + col];
}

// ---------------- WMMA GEMM (2x2 register-tiled) ----------------
// C[M,Ncols] (+bias, opt ReLU) = A[M,K](bf16, row-major) @ Bp(packed bf16)
// grid = (M/32, Ncols/256); block = 256 (8 waves). Each wave computes a
// 32x32 output patch: 2 A frags + 2 B frags -> 4 WMMAs per k-step
// (2 b128 loads per WMMA, 4 independent accumulator chains).

__global__ __launch_bounds__(256, 2) void gemm_bf16_wmma(
    const __bf16* __restrict__ A, const __bf16* __restrict__ Bp,
    const float* __restrict__ bias, float* __restrict__ C,
    __bf16* __restrict__ Cbf, int K, int Ncols, int relu) {
  const int lane   = threadIdx.x & 31;
  const int wave   = threadIdx.x >> 5;
  const int laneHi = lane >> 4;
  const int laneLo = lane & 15;
  const int m0 = blockIdx.x << 5;                 // 32 rows per block
  const int n0 = (blockIdx.y << 8) + (wave << 5); // 32 cols per wave

  // A fragments (16x32 bf16): lanes 0-15 -> K [0..15];
  // lanes 16-31 -> K [16..31]
  const __bf16* arow0 = A + (size_t)(m0 + laneLo) * K + (laneHi ? 16 : 0);
  const __bf16* arow1 = arow0 + (size_t)16 * K;
  const size_t  kt512 = (size_t)(K >> 5) * 512;
  const __bf16* bbase0 = Bp + (size_t)(n0 >> 4) * kt512 + (size_t)lane * 16;
  const __bf16* bbase1 = bbase0 + kt512;

  v8f acc00 = {0.f, 0.f, 0.f, 0.f, 0.f, 0.f, 0.f, 0.f};
  v8f acc01 = acc00, acc10 = acc00, acc11 = acc00;

#pragma unroll 2
  for (int k0 = 0; k0 < K; k0 += 32) {
    ABfrag a0, a1, b0, b1;
    a0.h[0] = *(const v8bf*)(arow0 + k0);
    a0.h[1] = *(const v8bf*)(arow0 + k0 + 8);
    a1.h[0] = *(const v8bf*)(arow1 + k0);
    a1.h[1] = *(const v8bf*)(arow1 + k0 + 8);
    const size_t boff = (size_t)(k0 >> 5) * 512;
    b0.h[0] = *(const v8bf*)(bbase0 + boff);
    b0.h[1] = *(const v8bf*)(bbase0 + boff + 8);
    b1.h[0] = *(const v8bf*)(bbase1 + boff);
    b1.h[1] = *(const v8bf*)(bbase1 + boff + 8);
    __builtin_prefetch(arow0 + k0 + 32, 0, 1);   // global_prefetch_b8
    __builtin_prefetch(arow1 + k0 + 32, 0, 1);
    acc00 = __builtin_amdgcn_wmma_f32_16x16x32_bf16(
        false, a0.v, false, b0.v, (short)0, acc00, false, false);
    acc01 = __builtin_amdgcn_wmma_f32_16x16x32_bf16(
        false, a0.v, false, b1.v, (short)0, acc01, false, false);
    acc10 = __builtin_amdgcn_wmma_f32_16x16x32_bf16(
        false, a1.v, false, b0.v, (short)0, acc10, false, false);
    acc11 = __builtin_amdgcn_wmma_f32_16x16x32_bf16(
        false, a1.v, false, b1.v, (short)0, acc11, false, false);
  }

  // C/D layout: VGPR r -> M = mtile + r + 8*laneHi ; N = ntile + laneLo
  const int col0 = n0 + laneLo;
  const int col1 = col0 + 16;
  const float bv0 = bias ? bias[col0] : 0.0f;
  const float bv1 = bias ? bias[col1] : 0.0f;
#pragma unroll
  for (int r2 = 0; r2 < 8; ++r2) {
    int rr0 = m0 + r2 + (laneHi << 3);
    int rr1 = rr0 + 16;
    float v00 = acc00[r2] + bv0, v01 = acc01[r2] + bv1;
    float v10 = acc10[r2] + bv0, v11 = acc11[r2] + bv1;
    if (relu) {
      v00 = v00 > 0.f ? v00 : 0.f;  v01 = v01 > 0.f ? v01 : 0.f;
      v10 = v10 > 0.f ? v10 : 0.f;  v11 = v11 > 0.f ? v11 : 0.f;
    }
    size_t i00 = (size_t)rr0 * Ncols + col0, i01 = (size_t)rr0 * Ncols + col1;
    size_t i10 = (size_t)rr1 * Ncols + col0, i11 = (size_t)rr1 * Ncols + col1;
    if (C) {
      C[i00] = v00; C[i01] = v01; C[i10] = v10; C[i11] = v11;
    }
    if (Cbf) {
      Cbf[i00] = (__bf16)v00; Cbf[i01] = (__bf16)v01;
      Cbf[i10] = (__bf16)v10; Cbf[i11] = (__bf16)v11;
    }
  }
}

// ---------------- edge attention (gather / score / scatter) ----------------
// One wave per edge. lane -> (head = lane>>2, octet = lane&3): each lane owns
// 8 consecutive dk values, so the wave reads the full 256-float rows of
// K[src], Q[dst], V[src] fully coalesced. 4-lane shuffle reduce per head dot.
// Q/K/V/WV are each 20.5 MB -> resident in the 192 MB L2, so the random
// per-edge gathers and float atomics are L2-bandwidth bound, not HBM bound.

__global__ __launch_bounds__(256) void edge_attn_kernel(
    const float* __restrict__ Q, const float* __restrict__ Kb,
    const float* __restrict__ Vb, const float* __restrict__ rel,
    const int* __restrict__ ef, const int* __restrict__ srcI,
    const int* __restrict__ dstI, float* __restrict__ WV,
    float* __restrict__ Z, int E) {
  int wave = threadIdx.x >> 5;
  int lane = threadIdx.x & 31;
  int e = blockIdx.x * 8 + wave;
  if (e >= E) return;
  int s = srcI[e], d = dstI[e], rf = ef[e];
  int h = lane >> 2, part = lane & 3;
  int off = h * DKH + part * 8;

  const float4* kp = (const float4*)(Kb  + (size_t)s * HSZ + off);
  const float4* qp = (const float4*)(Q   + (size_t)d * HSZ + off);
  const float4* vp = (const float4*)(Vb  + (size_t)s * HSZ + off);
  const float4* ep = (const float4*)(rel + (size_t)rf * DKH + part * 8);

  float4 k0 = kp[0], k1 = kp[1];
  float4 q0 = qp[0], q1 = qp[1];
  float4 e0 = ep[0], e1 = ep[1];

  float dot = (k0.x + e0.x) * q0.x + (k0.y + e0.y) * q0.y +
              (k0.z + e0.z) * q0.z + (k0.w + e0.w) * q0.w +
              (k1.x + e1.x) * q1.x + (k1.y + e1.y) * q1.y +
              (k1.z + e1.z) * q1.z + (k1.w + e1.w) * q1.w;
  dot += __shfl_xor(dot, 1, 32);
  dot += __shfl_xor(dot, 2, 32);

  const float inv_scale = 0.17677669529663687f;  // 1/sqrt(32)
  float score = __expf(fminf(fmaxf(dot * inv_scale, -10.0f), 10.0f));

  float4 v0 = vp[0], v1 = vp[1];
  float* wvp = WV + (size_t)d * HSZ + off;
  __hip_atomic_fetch_add(wvp + 0, (v0.x + e0.x) * score, __ATOMIC_RELAXED, __HIP_MEMORY_SCOPE_AGENT);
  __hip_atomic_fetch_add(wvp + 1, (v0.y + e0.y) * score, __ATOMIC_RELAXED, __HIP_MEMORY_SCOPE_AGENT);
  __hip_atomic_fetch_add(wvp + 2, (v0.z + e0.z) * score, __ATOMIC_RELAXED, __HIP_MEMORY_SCOPE_AGENT);
  __hip_atomic_fetch_add(wvp + 3, (v0.w + e0.w) * score, __ATOMIC_RELAXED, __HIP_MEMORY_SCOPE_AGENT);
  __hip_atomic_fetch_add(wvp + 4, (v1.x + e1.x) * score, __ATOMIC_RELAXED, __HIP_MEMORY_SCOPE_AGENT);
  __hip_atomic_fetch_add(wvp + 5, (v1.y + e1.y) * score, __ATOMIC_RELAXED, __HIP_MEMORY_SCOPE_AGENT);
  __hip_atomic_fetch_add(wvp + 6, (v1.z + e1.z) * score, __ATOMIC_RELAXED, __HIP_MEMORY_SCOPE_AGENT);
  __hip_atomic_fetch_add(wvp + 7, (v1.w + e1.w) * score, __ATOMIC_RELAXED, __HIP_MEMORY_SCOPE_AGENT);
  if (part == 0)
    __hip_atomic_fetch_add(Z + (size_t)d * NH + h, score, __ATOMIC_RELAXED, __HIP_MEMORY_SCOPE_AGENT);
}

// o = wv / z, emitted as bf16 for the O-projection GEMM
__global__ __launch_bounds__(256) void attn_norm_kernel(
    const float* __restrict__ WV, const float* __restrict__ Z,
    __bf16* __restrict__ Obf, int total) {
  int i = blockIdx.x * 256 + threadIdx.x;
  if (i < total) {
    int node = i >> 8;
    int c = i & 255;
    float z = Z[node * NH + (c >> 5)];
    Obf[i] = (__bf16)(WV[i] / z);
  }
}

// LayerNorm(A + B) * g + b, writing both f32 and bf16. One wave per row.
__global__ __launch_bounds__(256) void ln_kernel(
    const float* __restrict__ Aa, const float* __restrict__ Bb,
    const float* __restrict__ g, const float* __restrict__ bta,
    float* __restrict__ outF, __bf16* __restrict__ outBf, int Nrows) {
  int wave = threadIdx.x >> 5, lane = threadIdx.x & 31;
  int rowi = blockIdx.x * 8 + wave;
  if (rowi >= Nrows) return;
  const float* ap = Aa + (size_t)rowi * HSZ + lane * 8;
  const float* bp = Bb + (size_t)rowi * HSZ + lane * 8;
  float v[8];
  float sum = 0.0f;
#pragma unroll
  for (int j = 0; j < 8; ++j) { v[j] = ap[j] + bp[j]; sum += v[j]; }
  for (int o = 16; o > 0; o >>= 1) sum += __shfl_xor(sum, o, 32);
  float mean = sum * (1.0f / 256.0f);
  float var = 0.0f;
#pragma unroll
  for (int j = 0; j < 8; ++j) { float dd = v[j] - mean; var += dd * dd; }
  for (int o = 16; o > 0; o >>= 1) var += __shfl_xor(var, o, 32);
  float rstd = rsqrtf(var * (1.0f / 256.0f) + 1e-5f);
#pragma unroll
  for (int j = 0; j < 8; ++j) {
    int c = lane * 8 + j;
    float o_ = (v[j] - mean) * rstd * g[c] + bta[c];
    size_t idx = (size_t)rowi * HSZ + c;
    outF[idx]  = o_;
    outBf[idx] = (__bf16)o_;
  }
}

// ---------------- orchestration ----------------

extern "C" void kernel_launch(void* const* d_in, const int* in_sizes, int n_in,
                              void* d_out, int out_size, void* d_ws, size_t ws_size,
                              hipStream_t stream) {
  const float* x0   = (const float*)d_in[0];
  const int*   ef   = (const int*)d_in[1];
  const int*   srcI = (const int*)d_in[2];
  const int*   dstI = (const int*)d_in[3];
  const float* rel  = (const float*)d_in[4];
  const float* Wq   = (const float*)d_in[5];
  const float* bq   = (const float*)d_in[6];
  const float* Wk   = (const float*)d_in[7];
  const float* Wv   = (const float*)d_in[8];
  const float* Wo   = (const float*)d_in[9];
  const float* bo   = (const float*)d_in[10];
  const float* W1   = (const float*)d_in[11];
  const float* b1   = (const float*)d_in[12];
  const float* W2   = (const float*)d_in[13];
  const float* b2   = (const float*)d_in[14];
  const float* ln1g = (const float*)d_in[15];
  const float* ln1b = (const float*)d_in[16];
  const float* ln2g = (const float*)d_in[17];
  const float* ln2b = (const float*)d_in[18];

  char* cur = (char*)d_ws;
  auto alloc = [&](size_t bytes) -> void* {
    void* p = (void*)cur;
    cur += (bytes + 255) & ~(size_t)255;
    return p;
  };

  __bf16* Xbf   = (__bf16*)alloc((size_t)NHS * 2);
  float*  Qb    = (float*)alloc((size_t)NHS * 4);
  float*  Kbuf  = (float*)alloc((size_t)NHS * 4);
  float*  Vbuf  = (float*)alloc((size_t)NHS * 4);
  float*  WVb   = (float*)alloc((size_t)NHS * 4);
  float*  Zb    = (float*)alloc((size_t)NN * NH * 4);
  __bf16* Obf   = (__bf16*)alloc((size_t)NHS * 2);
  float*  T1    = (float*)alloc((size_t)NHS * 4);
  float*  Hf    = (float*)alloc((size_t)NHS * 4);
  __bf16* Hbf   = (__bf16*)alloc((size_t)NHS * 2);
  __bf16* FF1bf = (__bf16*)alloc((size_t)NN * 4 * HSZ * 2);
  float*  T2    = (float*)alloc((size_t)NHS * 4);
  float*  Xf    = (float*)alloc((size_t)NHS * 4);
  __bf16* WqP   = (__bf16*)alloc((size_t)NL * HSZ * HSZ * 2);
  __bf16* WkP   = (__bf16*)alloc((size_t)NL * HSZ * HSZ * 2);
  __bf16* WvP   = (__bf16*)alloc((size_t)NL * HSZ * HSZ * 2);
  __bf16* WoP   = (__bf16*)alloc((size_t)NL * HSZ * HSZ * 2);
  __bf16* W1P   = (__bf16*)alloc((size_t)NL * HSZ * 4 * HSZ * 2);
  __bf16* W2P   = (__bf16*)alloc((size_t)NL * 4 * HSZ * HSZ * 2);

  const int SQ = HSZ * HSZ;        // 65536
  const int SF = HSZ * 4 * HSZ;    // 262144

  // Pack all weights to bf16 fragment-major layout (tiny, once per call)
  for (int l = 0; l < NL; ++l) {
    pack_b_kernel<<<SQ / 256, 256, 0, stream>>>(Wq + (size_t)l * SQ, WqP + (size_t)l * SQ, HSZ, HSZ);
    pack_b_kernel<<<SQ / 256, 256, 0, stream>>>(Wk + (size_t)l * SQ, WkP + (size_t)l * SQ, HSZ, HSZ);
    pack_b_kernel<<<SQ / 256, 256, 0, stream>>>(Wv + (size_t)l * SQ, WvP + (size_t)l * SQ, HSZ, HSZ);
    pack_b_kernel<<<SQ / 256, 256, 0, stream>>>(Wo + (size_t)l * SQ, WoP + (size_t)l * SQ, HSZ, HSZ);
    pack_b_kernel<<<SF / 256, 256, 0, stream>>>(W1 + (size_t)l * SF, W1P + (size_t)l * SF, HSZ, 4 * HSZ);
    pack_b_kernel<<<SF / 256, 256, 0, stream>>>(W2 + (size_t)l * SF, W2P + (size_t)l * SF, 4 * HSZ, HSZ);
  }

  // Layer 0 input activations -> bf16 (subsequent layers: ln2 emits bf16)
  f32_to_bf16_kernel<<<NHS / 256, 256, 0, stream>>>(x0, Xbf, NHS);

  const float* xin = x0;
  dim3 g256(NN / 32, HSZ / 256);        // GEMMs with Ncols=256
  dim3 g1024(NN / 32, (4 * HSZ) / 256); // FFN1 (Ncols=1024)

  for (int l = 0; l < NL; ++l) {
    // Q/K/V projections (bf16 WMMA, fp32 accum)
    gemm_bf16_wmma<<<g256, 256, 0, stream>>>(Xbf, WqP + (size_t)l * SQ, bq + l * HSZ, Qb, nullptr, HSZ, HSZ, 0);
    gemm_bf16_wmma<<<g256, 256, 0, stream>>>(Xbf, WkP + (size_t)l * SQ, nullptr, Kbuf, nullptr, HSZ, HSZ, 0);
    gemm_bf16_wmma<<<g256, 256, 0, stream>>>(Xbf, WvP + (size_t)l * SQ, nullptr, Vbuf, nullptr, HSZ, HSZ, 0);

    // Edge attention: zero accumulators, gather/score/scatter (L2-resident)
    zero_kernel<<<NHS / 256, 256, 0, stream>>>(WVb, NHS);
    zero_kernel<<<(NN * NH + 255) / 256, 256, 0, stream>>>(Zb, NN * NH);
    edge_attn_kernel<<<(EE + 7) / 8, 256, 0, stream>>>(Qb, Kbuf, Vbuf, rel, ef, srcI, dstI, WVb, Zb, EE);
    attn_norm_kernel<<<NHS / 256, 256, 0, stream>>>(WVb, Zb, Obf, NHS);

    // O projection + LN1 (residual with x)
    gemm_bf16_wmma<<<g256, 256, 0, stream>>>(Obf, WoP + (size_t)l * SQ, bo + l * HSZ, T1, nullptr, HSZ, HSZ, 0);
    ln_kernel<<<NN / 8, 256, 0, stream>>>(xin, T1, ln1g + l * HSZ, ln1b + l * HSZ, Hf, Hbf, NN);

    // FFN: relu(h@W1+b1) emitted directly as bf16, then @W2+b2
    gemm_bf16_wmma<<<g1024, 256, 0, stream>>>(Hbf, W1P + (size_t)l * SF, b1 + l * 4 * HSZ, nullptr, FF1bf, HSZ, 4 * HSZ, 1);
    gemm_bf16_wmma<<<g256, 256, 0, stream>>>(FF1bf, W2P + (size_t)l * SF, b2 + l * HSZ, T2, nullptr, 4 * HSZ, HSZ, 0);

    // LN2 (residual with h) -> next layer x (f32 + bf16)
    float* xoutF = (l == NL - 1) ? (float*)d_out : Xf;
    ln_kernel<<<NN / 8, 256, 0, stream>>>(Hf, T2, ln2g + l * HSZ, ln2b + l * HSZ, xoutF, Xbf, NN);
    xin = xoutF;
  }
}
